// MemoryEfficientVocabOutput_77902116814964
// MI455X (gfx1250) — compile-verified, matrix operand order checked by reading, not run on hardware
//
#include <hip/hip_runtime.h>
#include <hip/hip_bf16.h>
#include <stdint.h>

// ---------------------------------------------------------------------------
// MEVO forward for MI455X (gfx1250, wave32, WMMA).
// loss = sum_t [ log(sum_v exp(x_t.w_v)) stabilized by max  -  x_t.w_target ]
//
// Compute-bound (537 GFLOP vs ~165MB min traffic) -> V_WMMA_F32_16X16X32_BF16
// with fp32 accumulation. W converted to bf16 once (65.5MB < 192MB L2) so the
// 640 GEMM blocks stream it from L2, not HBM.
//
// Round-3 fix: round-2's register-staged double buffer spilled to scratch
// (acc 64 + frags 48 + state 32 + staging 32 VGPRs). Replace VGPR staging
// with CDNA5 async global->LDS copies (GLOBAL_LOAD_ASYNC_TO_LDS_B128,
// ASYNCcnt): zero staging registers, double-buffered LDS, one barrier per
// K-stage, loads for stage s+1 in flight during stage-s WMMA bursts.
// ---------------------------------------------------------------------------

typedef __attribute__((ext_vector_type(16))) __bf16 v16bf;
typedef __attribute__((ext_vector_type(8)))  float  v8f;

#define TOKENS  8192
#define DMODEL  1024
#define VOCAB   32000
#define MT      128                      // tokens per block
#define NT      128                      // vocab per chunk
#define KT      64                       // K per LDS stage
#define KSTAGES (DMODEL / KT)            // 16
#define VSPLIT  10                       // vocab splits (grid.y)
#define PSPLIT  (VSPLIT * 2)             // logical partial splits (2 N-halves)
#define NCHUNKS (VOCAB / (NT * VSPLIT))  // 25 chunks of 128 vocab per split
#define LDSW    72                       // 64 + 8 pad; 144B row = 16B aligned
#define LROWS   (MT + NT)                // 256 LDS rows per buffer (A then B)

// Trivially-constructible 16-byte chunk (HIP's uint4 has ctors -> bad in unions)
struct __align__(16) U128 { uint32_t a, b, c, d; };
union Frag { U128 u[2]; v16bf v; };

__device__ __forceinline__ uint16_t f32_to_bf16(float f) {
  uint32_t u = __float_as_uint(f);
  u += 0x7FFFu + ((u >> 16) & 1u);       // round-to-nearest-even
  return (uint16_t)(u >> 16);
}

// CDNA5 async copy: 16 bytes per lane, global -> LDS, tracked by ASYNCcnt.
// GVS addressing: mem = SGPR64_base + VGPR_i32_offset. VDST = LDS byte addr.
__device__ __forceinline__ void async_load_b128(uint32_t lds_byte_addr,
                                                const void* sbase,
                                                int gbyte_off) {
  asm volatile("global_load_async_to_lds_b128 %0, %1, %2"
               :
               : "v"(lds_byte_addr), "v"(gbyte_off), "s"(sbase)
               : "memory");
}

__device__ __forceinline__ void wait_asynccnt0() {
  asm volatile("s_wait_asynccnt 0x0" ::: "memory");
}

// ------------------------------ kernels ------------------------------------

__global__ void zero_out_kernel(float* out) { out[0] = 0.0f; }

// fp32 -> bf16, 8 elements per thread, fully vectorized (n divisible by 8).
__global__ void convert_f32_bf16_kernel(const float* __restrict__ src,
                                        uint16_t* __restrict__ dst, int n) {
  int i = (blockIdx.x * blockDim.x + threadIdx.x) * 8;
  if (i >= n) return;
  float4 lo = *(const float4*)(src + i);
  float4 hi = *(const float4*)(src + i + 4);
  union { uint16_t h[8]; U128 u; } o;
  o.h[0] = f32_to_bf16(lo.x); o.h[1] = f32_to_bf16(lo.y);
  o.h[2] = f32_to_bf16(lo.z); o.h[3] = f32_to_bf16(lo.w);
  o.h[4] = f32_to_bf16(hi.x); o.h[5] = f32_to_bf16(hi.y);
  o.h[6] = f32_to_bf16(hi.z); o.h[7] = f32_to_bf16(hi.w);
  *(U128*)(dst + i) = o.u;
}

// GEMM (bf16 WMMA, fp32 acc) + online softmax over the vocab slice.
// grid = (TOKENS/MT, VSPLIT), block = 256 threads = 8 waves in a 4(M)x2(N)
// arrangement: wave owns a 32-token x 64-vocab strip per 128x128 chunk.
__global__ __launch_bounds__(256)
void mevo_gemm_softmax_kernel(const uint16_t* __restrict__ xbf,
                              const uint16_t* __restrict__ wbf,
                              float* __restrict__ maxp,
                              float* __restrict__ sump) {
  // Double-buffered tile store: rows 0..127 = A (tokens), 128..255 = B (vocab).
  __shared__ __align__(16) uint16_t smem[2][LROWS][LDSW];

  const int tid  = threadIdx.x;
  const int wave = tid >> 5;       // 0..7
  const int mg   = wave >> 1;      // M group 0..3 -> M subtiles {2mg, 2mg+1}
  const int ng   = wave & 1;       // N group 0..1 -> N subtiles {4ng..4ng+3}
  const int lane = tid & 31;
  const int lh   = lane >> 4;      // half-wave: K 0..7 vs 8..15 frag pattern
  const int l16  = lane & 15;
  const int tokBase = blockIdx.x * MT;
  const int vSplit  = blockIdx.y * (NCHUNKS * NT);

  // LDS aperture keeps the LDS offset in addr[31:0] -> low dword of the flat
  // shared pointer is the byte address the async-copy VDST operand needs.
  const uint32_t lds0 = (uint32_t)(uintptr_t)&smem[0][0][0];

  // Per-thread staging coords: 256 threads x 4 iters x 16B cover one 128x64
  // bf16 tile; done once for A and once for B per stage (8 async ops/thread).
  const int sr = tid >> 3;         // row 0..31 (+32*i below)
  const int sc = (tid & 7) * 8;    // 8-elem column 0,8,..,56

  // Issue all async copies for pipeline stage s (chunk = s/16, kt = (s%16)*64).
  auto issueStage = [&](int s) {
    const int buf = s & 1;
    const int ktn = (s & (KSTAGES - 1)) * KT;
    const int vb  = vSplit + (s >> 4) * NT;
#pragma unroll
    for (int i = 0; i < 4; ++i) {
      const int r = sr + i * 32;
      const uint32_t la = lds0 + (uint32_t)(((buf * LROWS + r) * LDSW + sc) * 2);
      const uint32_t lb = lds0 + (uint32_t)(((buf * LROWS + MT + r) * LDSW + sc) * 2);
      const int ga = ((tokBase + r) * DMODEL + ktn + sc) * 2;
      const int gb = ((vb + r) * DMODEL + ktn + sc) * 2;
      async_load_b128(la, xbf, ga);
      async_load_b128(lb, wbf, gb);
    }
  };

  // Running online-softmax state for 16 rows: [m-subtile][C-vgpr].
  // C layout: VGPR v holds rows M=v (lanes 0-15) and M=v+8 (lanes 16-31).
  float rm[2][8], rs[2][8];
#pragma unroll
  for (int m = 0; m < 2; ++m)
#pragma unroll
    for (int v = 0; v < 8; ++v) { rm[m][v] = -3.0e38f; rs[m][v] = 0.0f; }

  const int TOTAL = NCHUNKS * KSTAGES;
  issueStage(0);

  for (int chunk = 0; chunk < NCHUNKS; ++chunk) {
    v8f acc[2][4];
#pragma unroll
    for (int m = 0; m < 2; ++m)
#pragma unroll
      for (int n = 0; n < 4; ++n) acc[m][n] = (v8f){0,0,0,0,0,0,0,0};

    for (int kq = 0; kq < KSTAGES; ++kq) {
      const int s   = chunk * KSTAGES + kq;
      const int buf = s & 1;

      wait_asynccnt0();      // own stage-s copies have landed in LDS
      __syncthreads();       // all waves landed; prev stage's compute done
      if (s + 1 < TOTAL) issueStage(s + 1);   // flies during the WMMA bursts

#pragma unroll
      for (int kk = 0; kk < KT; kk += 32) {
        // CDNA5 16-bit frag layout (16x32): lanes 0-15 take K=kb..kb+7 in
        // VGPR0-3, K=kb+16..kb+23 in VGPR4-7 (kb=kk); lanes 16-31 use kb=kk+8.
        const int kb = kk + lh * 8;
        Frag fa[2], fb[4];
#pragma unroll
        for (int m = 0; m < 2; ++m) {
          const uint16_t* p = &smem[buf][(mg * 2 + m) * 16 + l16][kb];
          fa[m].u[0] = *(const U128*)p;
          fa[m].u[1] = *(const U128*)(p + 16);
        }
#pragma unroll
        for (int n = 0; n < 4; ++n) {
          const uint16_t* p = &smem[buf][MT + (ng * 4 + n) * 16 + l16][kb];
          fb[n].u[0] = *(const U128*)p;
          fb[n].u[1] = *(const U128*)(p + 16);
        }
        // 8 WMMAs back-to-back off one dscnt wait.
#pragma unroll
        for (int m = 0; m < 2; ++m)
#pragma unroll
          for (int n = 0; n < 4; ++n)
            acc[m][n] = __builtin_amdgcn_wmma_f32_16x16x32_bf16(
                false, fa[m].v, false, fb[n].v, (short)0, acc[m][n], false, false);
      }
    }

    // Online-softmax fold of the 32x64 strip (registers only; overlaps the
    // already-issued async copies for the next chunk). xor masks 1,2,4,8 stay
    // inside each 16-lane half, matching the C-matrix N=lane layout.
#pragma unroll
    for (int m = 0; m < 2; ++m) {
#pragma unroll
      for (int v = 0; v < 8; ++v) {
        float m_old = rm[m][v];
        float m_new = m_old;
#pragma unroll
        for (int n = 0; n < 4; ++n) {
          float t = acc[m][n][v];
          t = fmaxf(t, __shfl_xor(t, 1));
          t = fmaxf(t, __shfl_xor(t, 2));
          t = fmaxf(t, __shfl_xor(t, 4));
          t = fmaxf(t, __shfl_xor(t, 8));
          m_new = fmaxf(m_new, t);
        }
        float s = rs[m][v] * __expf(m_old - m_new);
#pragma unroll
        for (int n = 0; n < 4; ++n) {
          float e = __expf(acc[m][n][v] - m_new);
          e += __shfl_xor(e, 1);
          e += __shfl_xor(e, 2);
          e += __shfl_xor(e, 4);
          e += __shfl_xor(e, 8);
          s += e;
        }
        rm[m][v] = m_new;
        rs[m][v] = s;
      }
    }
  }

  // One lane per half writes its 8 rows per m-subtile. Partial-split index
  // folds in the N-half owner; each (split, token) written by exactly one wave.
  if (l16 == 0) {
    const size_t ps = (size_t)(blockIdx.y * 2 + ng) * TOKENS;
#pragma unroll
    for (int m = 0; m < 2; ++m) {
#pragma unroll
      for (int v = 0; v < 8; ++v) {
        int token = tokBase + (mg * 2 + m) * 16 + lh * 8 + v;
        maxp[ps + token] = rm[m][v];
        sump[ps + token] = rs[m][v];
      }
    }
  }
}

// Merge the PSPLIT partial (max, expsum) pairs per token -> log S + M.
__global__ void mevo_reduce_kernel(const float* __restrict__ maxp,
                                   const float* __restrict__ sump,
                                   float* out) {
  int t = blockIdx.x * blockDim.x + threadIdx.x;
  if (t >= TOKENS) return;
  float M = -3.0e38f;
#pragma unroll
  for (int i = 0; i < PSPLIT; ++i) M = fmaxf(M, maxp[(size_t)i * TOKENS + t]);
  float S = 0.0f;
#pragma unroll
  for (int i = 0; i < PSPLIT; ++i)
    S += sump[(size_t)i * TOKENS + t] * __expf(maxp[(size_t)i * TOKENS + t] - M);
  atomicAdd(out, __logf(S) + M);
}

// Target score: one wave per token, fp32 dot(x_t, W[target_t]) (memory-bound,
// ~67MB total; keep full fp32 per the fp_target path).
__global__ void mevo_target_kernel(const float* __restrict__ x,
                                   const float* __restrict__ w,
                                   const int* __restrict__ tgt,
                                   float* out) {
  int token = blockIdx.x * 8 + (threadIdx.x >> 5);
  int lane  = threadIdx.x & 31;
  const float* xr = x + (size_t)token * DMODEL;
  const float* wr = w + (size_t)tgt[token] * DMODEL;
  float acc = 0.0f;
#pragma unroll
  for (int i = 0; i < 8; ++i) {
    int c = (lane + 32 * i) * 4;             // 32 lanes * 8 float4 = 1024
    float4 a = *(const float4*)(xr + c);
    float4 b = *(const float4*)(wr + c);
    acc = fmaf(a.x, b.x, acc);
    acc = fmaf(a.y, b.y, acc);
    acc = fmaf(a.z, b.z, acc);
    acc = fmaf(a.w, b.w, acc);
  }
  acc += __shfl_xor(acc, 1);
  acc += __shfl_xor(acc, 2);
  acc += __shfl_xor(acc, 4);
  acc += __shfl_xor(acc, 8);
  acc += __shfl_xor(acc, 16);
  if (lane == 0) atomicAdd(out, -acc);
}

// ------------------------------ launcher -----------------------------------

extern "C" void kernel_launch(void* const* d_in, const int* in_sizes, int n_in,
                              void* d_out, int out_size, void* d_ws, size_t ws_size,
                              hipStream_t stream) {
  const float* x   = (const float*)d_in[0];   // [8192, 1024] fp32
  const float* w   = (const float*)d_in[1];   // [32000, 1024] fp32
  const int*   tgt = (const int*)d_in[2];     // [8192] int32
  float*       out = (float*)d_out;           // scalar fp32

  // Workspace: x_bf16 (16MB) | w_bf16 (65.5MB) | maxp | sump (2 x 640KB)
  uint16_t* xbf  = (uint16_t*)d_ws;
  uint16_t* wbf  = xbf + (size_t)TOKENS * DMODEL;
  float*    maxp = (float*)(wbf + (size_t)VOCAB * DMODEL);
  float*    sump = maxp + (size_t)PSPLIT * TOKENS;

  zero_out_kernel<<<1, 1, 0, stream>>>(out);

  convert_f32_bf16_kernel<<<(TOKENS * DMODEL / 8 + 255) / 256, 256, 0, stream>>>(
      x, xbf, TOKENS * DMODEL);
  convert_f32_bf16_kernel<<<(VOCAB * DMODEL / 8 + 255) / 256, 256, 0, stream>>>(
      w, wbf, VOCAB * DMODEL);

  dim3 grid(TOKENS / MT, VSPLIT);   // 64 x 10 = 640 blocks, 8 waves each
  mevo_gemm_softmax_kernel<<<grid, 256, 0, stream>>>(xbf, wbf, maxp, sump);

  mevo_reduce_kernel<<<(TOKENS + 255) / 256, 256, 0, stream>>>(maxp, sump, out);
  mevo_target_kernel<<<TOKENS / 8, 256, 0, stream>>>(x, w, tgt, out);
}
